// BERTAttention_67095979098738
// MI455X (gfx1250) — matrix operand order repaired and checked by reading, not compiled, not run on hardware
//
#include <hip/hip_runtime.h>
#include <hip/hip_bf16.h>

typedef __bf16 bf16_t;
typedef __attribute__((ext_vector_type(4)))  float   v4f;
typedef __attribute__((ext_vector_type(8)))  float   v8f;
typedef __attribute__((ext_vector_type(4)))  __bf16  v4bf;
typedef __attribute__((ext_vector_type(8)))  __bf16  v8bf;
typedef __attribute__((ext_vector_type(16))) __bf16  v16bf;

#define D_MODEL 1024
#define N_HEADS 16
#define DK      64
#define BATCH   4
#define SEQ     2048
#define M_TOTAL (BATCH * SEQ)   // 8192 rows

// ---------------------------------------------------------------------------
// Fragment helpers (CDNA5 WMMA bf16 16x16x32 layouts, ISA 7.12.2)
// A (16x32, MxK): lane L -> row m = L%16; elems 0..7 -> k = h*8+0..7,
//                 elems 8..15 -> k = h*8+16..23  (h = L/16)
// B (32x16, KxN): lane L -> col n = L%16; elems 0..7 -> k = h*8+0..7,
//                 elems 8..15 -> k = h*8+16..23
// ---------------------------------------------------------------------------
static __device__ __forceinline__ v16bf cat8(v8bf lo, v8bf hi) {
  return __builtin_shufflevector(lo, hi, 0,1,2,3,4,5,6,7,8,9,10,11,12,13,14,15);
}
static __device__ __forceinline__ v16bf load_a_frag(const bf16_t* row_k0, int half) {
  const v8bf* p = reinterpret_cast<const v8bf*>(row_k0 + half * 8);
  return cat8(p[0], p[2]);                     // k = h*8..+7 and h*8+16..+23
}
static __device__ __forceinline__ v16bf load_b_frag(const bf16_t* col_k0, int half) {
  const v8bf* p = reinterpret_cast<const v8bf*>(col_k0 + half * 8);
  return cat8(p[0], p[2]);                     // k = h*8..+7 and h*8+16..+23
}
static __device__ __forceinline__ v8f wmma_bf16(v16bf a, v16bf b, v8f c) {
  return __builtin_amdgcn_wmma_f32_16x16x32_bf16(false, a, false, b, (short)0, c,
                                                 false, false);
}

// ---------------------------------------------------------------------------
// Kernel 1: fp32 -> bf16 convert (8-wide: two b128 loads -> one b128 store)
// ---------------------------------------------------------------------------
__global__ __launch_bounds__(256) void k_cvt8(const float4* __restrict__ in,
                                              v8bf* __restrict__ out, int n8) {
  int i = blockIdx.x * 256 + threadIdx.x;
  if (i < n8) {
    float4 f0 = in[2 * i], f1 = in[2 * i + 1];
    v8bf o;
    o[0] = (bf16_t)f0.x; o[1] = (bf16_t)f0.y; o[2] = (bf16_t)f0.z; o[3] = (bf16_t)f0.w;
    o[4] = (bf16_t)f1.x; o[5] = (bf16_t)f1.y; o[6] = (bf16_t)f1.z; o[7] = (bf16_t)f1.w;
    out[i] = o;
  }
}

// ---------------------------------------------------------------------------
// Shared 64x64-per-wave GEMM core, software-pipelined (2-stage double
// buffer, k-loop unrolled by 2 so each fragment tile's loads are issued a
// full 16-WMMA tile ahead of consumption).  Requires K % 64 == 0.
// ---------------------------------------------------------------------------
static __device__ __forceinline__ void load_frags(
    const bf16_t* const arow[4], const bf16_t* const brow[4], int kb, int half,
    v16bf a[4], v16bf b[4]) {
#pragma unroll
  for (int i = 0; i < 4; ++i) a[i] = load_a_frag(arow[i] + kb, half);
#pragma unroll
  for (int j = 0; j < 4; ++j) b[j] = load_b_frag(brow[j] + kb, half);
}
static __device__ __forceinline__ void wmma_tile(const v16bf a[4],
                                                 const v16bf b[4],
                                                 v8f acc[4][4]) {
#pragma unroll
  for (int i = 0; i < 4; ++i)
#pragma unroll
    for (int j = 0; j < 4; ++j) acc[i][j] = wmma_bf16(a[i], b[j], acc[i][j]);
}

static __device__ __forceinline__ void gemm64x64(
    const bf16_t* __restrict__ Abase, const bf16_t* __restrict__ Bbase,
    int K, int lm, int half, v8f acc[4][4]) {
  const bf16_t* arow[4];
  const bf16_t* brow[4];
#pragma unroll
  for (int i = 0; i < 4; ++i) arow[i] = Abase + (size_t)(i * 16 + lm) * K;
#pragma unroll
  for (int j = 0; j < 4; ++j) brow[j] = Bbase + (size_t)(j * 16 + lm) * K;

  v16bf a0[4], b0[4], a1[4], b1[4];
  load_frags(arow, brow, 0, half, a0, b0);
  for (int kb = 0; kb < K - 64; kb += 64) {
    load_frags(arow, brow, kb + 32, half, a1, b1);
    __builtin_prefetch(arow[0] + kb + 128, 0, 0);
    wmma_tile(a0, b0, acc);
    load_frags(arow, brow, kb + 64, half, a0, b0);
    __builtin_prefetch(brow[0] + kb + 128, 0, 0);
    wmma_tile(a1, b1, acc);
  }
  load_frags(arow, brow, K - 32, half, a1, b1);
  wmma_tile(a0, b0, acc);
  wmma_tile(a1, b1, acc);
}

// ---------------------------------------------------------------------------
// Kernel 2: fused QKV projection.  grid = (M/256, 1024/128, 3), block = 256
// (8 waves as 4x2, wave tile 64x64).  Q scaled by 1/sqrt(DK) at write time;
// Q,K stored (b,h,s,d); V stored transposed (b,h,d,s).
// ---------------------------------------------------------------------------
__global__ __launch_bounds__(256) void k_gemm_qkv(
    const bf16_t* __restrict__ X, const bf16_t* __restrict__ W3,
    const float* __restrict__ bq, const float* __restrict__ bk,
    const float* __restrict__ bv,
    bf16_t* __restrict__ Q, bf16_t* __restrict__ Kd, bf16_t* __restrict__ Vt) {
  const int tid = threadIdx.x;
  const int wave = tid >> 5, lane = tid & 31;
  const int lm = lane & 15, half = lane >> 4;
  const int proj = blockIdx.z;
  const bf16_t* W = W3 + (size_t)proj * D_MODEL * D_MODEL;

  const int m0 = blockIdx.x * 256 + (wave >> 1) * 64;
  const int n0 = blockIdx.y * 128 + (wave & 1) * 64;

  v8f acc[4][4] = {};
  gemm64x64(X + (size_t)m0 * D_MODEL, W + (size_t)n0 * D_MODEL, D_MODEL,
            lm, half, acc);

  // Uniform branch on proj (blockIdx.z) -- hoisted out of the store loops.
  if (proj == 2) {                     // V: transposed (b,h,d,s)
#pragma unroll
    for (int j = 0; j < 4; ++j) {
      const int nn = n0 + j * 16 + lm;
      const float bnn = bv[nn];
      const int h = nn >> 6, d = nn & 63;
#pragma unroll
      for (int i = 0; i < 4; ++i)
#pragma unroll
        for (int r = 0; r < 8; ++r) {
          const int mm = m0 + i * 16 + r + half * 8;
          const int bb = mm >> 11, ss = mm & 2047;
          const size_t bh = (size_t)bb * N_HEADS + h;
          Vt[(bh * DK + d) * SEQ + ss] = (bf16_t)(acc[i][j][r] + bnn);
        }
    }
  } else {                             // Q or K: (b,h,s,d); Q pre-scaled
    bf16_t* dst = (proj == 0) ? Q : Kd;
    const float* bias = (proj == 0) ? bq : bk;
    const float scale = (proj == 0) ? 0.125f : 1.0f;   // 1/sqrt(64)
#pragma unroll
    for (int j = 0; j < 4; ++j) {
      const int nn = n0 + j * 16 + lm;
      const float bnn = bias[nn];
      const int h = nn >> 6, d = nn & 63;
#pragma unroll
      for (int i = 0; i < 4; ++i)
#pragma unroll
        for (int r = 0; r < 8; ++r) {
          const int mm = m0 + i * 16 + r + half * 8;
          const int bb = mm >> 11, ss = mm & 2047;
          const size_t bh = (size_t)bb * N_HEADS + h;
          dst[(bh * SEQ + ss) * DK + d] = (bf16_t)((acc[i][j][r] + bnn) * scale);
        }
    }
  }
}

// ---------------------------------------------------------------------------
// Kernel 3: flash attention.  grid = (SEQ/128, B*H), block = 128 (4 waves).
// Each wave owns 32 query rows (2 M-tiles); streams 32-key tiles.
// Softmax row pass uses ALL 32 lanes (one row per lane).  Mask applied per
// column before the LDS spill; scores arrive pre-scaled (Q carries 1/8).
// ---------------------------------------------------------------------------
__global__ __launch_bounds__(128) void k_attn(
    const bf16_t* __restrict__ Q, const bf16_t* __restrict__ Kd,
    const bf16_t* __restrict__ Vt, const int* __restrict__ mask,
    bf16_t* __restrict__ Ctx) {
  const int tid = threadIdx.x;
  const int wave = tid >> 5, lane = tid & 31;
  const int lm = lane & 15, half = lane >> 4;
  const int bh = blockIdx.y, b = bh >> 4, h = bh & 15;
  const int q0 = blockIdx.x * 128 + wave * 32;
  const float NEG_INF = -__builtin_inff();

  const bf16_t* Qb = Q + (size_t)bh * SEQ * DK;
  const bf16_t* Kb = Kd + (size_t)bh * SEQ * DK;
  const bf16_t* Vb = Vt + (size_t)bh * DK * SEQ;
  const int* mrow = mask + b * SEQ;

  __shared__ __align__(16) float  sS[4][32][32];   // score tiles (per wave)
  __shared__ __align__(16) bf16_t sP[4][32][32];   // exp() tiles, bf16
  __shared__ float sF[4][32];                      // per-row broadcast factors

  v16bf qA[2][2];
#pragma unroll
  for (int mi = 0; mi < 2; ++mi)
#pragma unroll
    for (int dh = 0; dh < 2; ++dh)
      qA[mi][dh] = load_a_frag(Qb + (size_t)(q0 + mi * 16 + lm) * DK + dh * 32, half);

  float mR = NEG_INF, lR = 0.f;
  v8f o[2][4] = {};

  for (int kt = 0; kt < SEQ; kt += 32) {
    // Keep the next key/value tiles flowing into WGP$ across the softmax
    // phase (sequential streams; speculative prefetch is free).
    __builtin_prefetch(Kb + (size_t)(kt + 32 + lm) * DK, 0, 0);
    __builtin_prefetch(Vb + (size_t)lm * SEQ + kt + 32, 0, 0);

    // --- S tiles: 32 q x 32 keys, K-dim 64; K-frags shared across M-tiles ---
    v8f sc[2][2] = {};
#pragma unroll
    for (int j = 0; j < 2; ++j)
#pragma unroll
      for (int dh = 0; dh < 2; ++dh) {
        v16bf kB = load_b_frag(Kb + (size_t)(kt + j * 16 + lm) * DK + dh * 32, half);
#pragma unroll
        for (int mi = 0; mi < 2; ++mi)
          sc[mi][j] = wmma_bf16(qA[mi][dh], kB, sc[mi][j]);
      }
    // Column mask: this lane owns key columns kt+lm and kt+16+lm.
    const float pen0 = mrow[kt + lm]      ? 0.f : NEG_INF;
    const float pen1 = mrow[kt + 16 + lm] ? 0.f : NEG_INF;
#pragma unroll
    for (int mi = 0; mi < 2; ++mi)
#pragma unroll
      for (int r = 0; r < 8; ++r) {
        sS[wave][mi * 16 + r + half * 8][lm]      = sc[mi][0][r] + pen0;
        sS[wave][mi * 16 + r + half * 8][16 + lm] = sc[mi][1][r] + pen1;
      }
    __syncthreads();

    // --- online softmax: every lane owns one of the wave's 32 rows ---
    {
      const int m = lane;
      v4f vv[8];
      const v4f* srow = reinterpret_cast<const v4f*>(&sS[wave][m][0]);
#pragma unroll
      for (int q = 0; q < 8; ++q) vv[q] = srow[q];
      float tmax = NEG_INF;
#pragma unroll
      for (int q = 0; q < 8; ++q)
#pragma unroll
        for (int e = 0; e < 4; ++e) tmax = fmaxf(tmax, vv[q][e]);
      const float mnew = fmaxf(mR, tmax);
      const bool dead = (mnew == NEG_INF);
      const float corr = dead ? 1.f : __expf(mR - mnew);
      float ssum = 0.f;
      v4bf* prow = reinterpret_cast<v4bf*>(&sP[wave][m][0]);
#pragma unroll
      for (int q = 0; q < 8; ++q) {
        v4bf pk;
#pragma unroll
        for (int e = 0; e < 4; ++e) {
          const float p = dead ? 0.f : __expf(vv[q][e] - mnew);
          ssum += p;
          pk[e] = (bf16_t)p;
        }
        prow[q] = pk;
      }
      lR = lR * corr + ssum;
      mR = mnew;
      sF[wave][m] = corr;
    }
    __syncthreads();

    // --- rescale running O, then O += P x V (V-frags shared across M) ---
    float cf[2][8];
#pragma unroll
    for (int mi = 0; mi < 2; ++mi)
#pragma unroll
      for (int r = 0; r < 8; ++r) cf[mi][r] = sF[wave][mi * 16 + r + half * 8];
#pragma unroll
    for (int mi = 0; mi < 2; ++mi)
#pragma unroll
      for (int t = 0; t < 4; ++t)
#pragma unroll
        for (int r = 0; r < 8; ++r) o[mi][t][r] *= cf[mi][r];

    v16bf pA[2];
#pragma unroll
    for (int mi = 0; mi < 2; ++mi) {
      const v8bf* pp = reinterpret_cast<const v8bf*>(&sP[wave][mi * 16 + lm][half * 8]);
      pA[mi] = cat8(pp[0], pp[2]);
    }
#pragma unroll
    for (int t = 0; t < 4; ++t) {
      v16bf vB = load_b_frag(Vb + (size_t)(t * 16 + lm) * SEQ + kt, half);
#pragma unroll
      for (int mi = 0; mi < 2; ++mi) o[mi][t] = wmma_bf16(pA[mi], vB, o[mi][t]);
    }
    __syncthreads();
  }

  sF[wave][lane] = (lR > 0.f) ? (1.f / lR) : 0.f;
  __syncthreads();
  float inv[2][8];
#pragma unroll
  for (int mi = 0; mi < 2; ++mi)
#pragma unroll
    for (int r = 0; r < 8; ++r) inv[mi][r] = sF[wave][mi * 16 + r + half * 8];
#pragma unroll
  for (int mi = 0; mi < 2; ++mi)
#pragma unroll
    for (int t = 0; t < 4; ++t)
#pragma unroll
      for (int r = 0; r < 8; ++r) {
        const int m = mi * 16 + r + half * 8;
        Ctx[(size_t)(b * SEQ + q0 + m) * D_MODEL + h * DK + t * 16 + lm] =
            (bf16_t)(o[mi][t][r] * inv[mi][r]);
      }
}

// ---------------------------------------------------------------------------
// Kernel 4: output projection.  out = ctx . Wo^T + bo  (fp32 result)
//   grid = (M/256, 1024/128), block = 256, wave tile 64x64.
// ---------------------------------------------------------------------------
__global__ __launch_bounds__(256) void k_gemm_out(
    const bf16_t* __restrict__ Ctx, const bf16_t* __restrict__ Wo,
    const float* __restrict__ bo, float* __restrict__ out) {
  const int tid = threadIdx.x;
  const int wave = tid >> 5, lane = tid & 31;
  const int lm = lane & 15, half = lane >> 4;

  const int m0 = blockIdx.x * 256 + (wave >> 1) * 64;
  const int n0 = blockIdx.y * 128 + (wave & 1) * 64;

  v8f acc[4][4] = {};
  gemm64x64(Ctx + (size_t)m0 * D_MODEL, Wo + (size_t)n0 * D_MODEL, D_MODEL,
            lm, half, acc);

#pragma unroll
  for (int j = 0; j < 4; ++j) {
    const int nn = n0 + j * 16 + lm;
    const float bnn = bo[nn];
#pragma unroll
    for (int i = 0; i < 4; ++i)
#pragma unroll
      for (int r = 0; r < 8; ++r) {
        const int mm = m0 + i * 16 + r + half * 8;
        out[(size_t)mm * D_MODEL + nn] = acc[i][j][r] + bnn;
      }
  }
}

// ---------------------------------------------------------------------------
// Host launcher
// ---------------------------------------------------------------------------
extern "C" void kernel_launch(void* const* d_in, const int* in_sizes, int n_in,
                              void* d_out, int out_size, void* d_ws, size_t ws_size,
                              hipStream_t stream) {
  const float* x   = (const float*)d_in[0];
  const int*  mask = (const int*)d_in[1];
  const float* Wq  = (const float*)d_in[2];
  const float* bq  = (const float*)d_in[3];
  const float* Wk  = (const float*)d_in[4];
  const float* bk  = (const float*)d_in[5];
  const float* Wv  = (const float*)d_in[6];
  const float* bv  = (const float*)d_in[7];
  const float* Wo  = (const float*)d_in[8];
  const float* bo  = (const float*)d_in[9];
  float* out = (float*)d_out;

  // Workspace carve-up (all bf16): X, Wq|Wk|Wv, Wo, Q, K, V^T, Ctx  (~88 MB)
  char* ws = (char*)d_ws;
  bf16_t* Xbf = (bf16_t*)ws;  ws += (size_t)M_TOTAL * D_MODEL * sizeof(bf16_t);
  bf16_t* W3  = (bf16_t*)ws;  ws += (size_t)3 * D_MODEL * D_MODEL * sizeof(bf16_t);
  bf16_t* Wob = (bf16_t*)ws;  ws += (size_t)D_MODEL * D_MODEL * sizeof(bf16_t);
  bf16_t* Qb  = (bf16_t*)ws;  ws += (size_t)M_TOTAL * D_MODEL * sizeof(bf16_t);
  bf16_t* Kb  = (bf16_t*)ws;  ws += (size_t)M_TOTAL * D_MODEL * sizeof(bf16_t);
  bf16_t* Vtb = (bf16_t*)ws;  ws += (size_t)M_TOTAL * D_MODEL * sizeof(bf16_t);
  bf16_t* Ctx = (bf16_t*)ws;

  const int nX = M_TOTAL * D_MODEL;       // 8388608
  const int nW = D_MODEL * D_MODEL;       // 1048576
  k_cvt8<<<nX / 8 / 256, 256, 0, stream>>>((const float4*)x,  (v8bf*)Xbf, nX / 8);
  k_cvt8<<<nW / 8 / 256, 256, 0, stream>>>((const float4*)Wq, (v8bf*)(W3 + 0 * (size_t)nW), nW / 8);
  k_cvt8<<<nW / 8 / 256, 256, 0, stream>>>((const float4*)Wk, (v8bf*)(W3 + 1 * (size_t)nW), nW / 8);
  k_cvt8<<<nW / 8 / 256, 256, 0, stream>>>((const float4*)Wv, (v8bf*)(W3 + 2 * (size_t)nW), nW / 8);
  k_cvt8<<<nW / 8 / 256, 256, 0, stream>>>((const float4*)Wo, (v8bf*)Wob, nW / 8);

  dim3 gQKV(M_TOTAL / 256, D_MODEL / 128, 3);
  k_gemm_qkv<<<gQKV, 256, 0, stream>>>(Xbf, W3, bq, bk, bv, Qb, Kb, Vtb);

  dim3 gAtt(SEQ / 128, BATCH * N_HEADS);
  k_attn<<<gAtt, 128, 0, stream>>>(Qb, Kb, Vtb, mask, Ctx);

  dim3 gOut(M_TOTAL / 256, D_MODEL / 128);
  k_gemm_out<<<gOut, 256, 0, stream>>>(Ctx, Wob, bo, out);
}
